// BatchMultiHeadGraphAttention_5901285065231
// MI455X (gfx1250) — matrix operand order repaired and checked by reading, not compiled, not run on hardware
//
#include <hip/hip_runtime.h>
#include <hip/hip_bf16.h>
#include <cstddef>

typedef __attribute__((ext_vector_type(16))) _Float16 v16h;
typedef __attribute__((ext_vector_type(8)))  _Float16 v8h;
typedef __attribute__((ext_vector_type(8)))  float    v8f;
typedef __attribute__((ext_vector_type(4)))  float    v4f;
typedef __attribute__((ext_vector_type(4)))  int      v4i;

#define BS   16
#define NH   4
#define NN   1024
#define FI   128
#define FO   64
#define NEG_SLOPE 0.2f

// ---------------------------------------------------------------------------
// k0: w [H][FI][FO] fp32  ->  w_t [H][FO][FI] f16 (transposed for B-frag loads)
// ---------------------------------------------------------------------------
__global__ void k0_convert_w(const float* __restrict__ w, _Float16* __restrict__ w_t) {
    int idx = blockIdx.x * blockDim.x + threadIdx.x;    // 0 .. NH*FO*FI-1 = 32767
    int hh  = idx >> 13;                                 // / (FO*FI)
    int rem = idx & 8191;
    int o   = rem >> 7;                                  // / FI
    int k   = rem & 127;
    w_t[idx] = (_Float16)w[((size_t)hh * FI + k) * FO + o];
}

// ---------------------------------------------------------------------------
// k1: h_prime[bh][n][o] = h[b] @ w[head]   (f16 WMMA, f32 accumulate)
// One 16x16 output tile per wave; K loop = 4 steps of 32.
// ---------------------------------------------------------------------------
__global__ __launch_bounds__(128) void k1_hprime(const float* __restrict__ h,
                                                 const _Float16* __restrict__ w_t,
                                                 float* __restrict__ hp) {
    int wave = threadIdx.x >> 5, lane = threadIdx.x & 31;
    int hi = lane >> 4, n16 = lane & 15;

    int tile    = blockIdx.x * 4 + wave;        // 0..16383
    int cc      = tile & 3;                      // column chunk (16 cols)
    int rowtile = (tile >> 2) & 63;
    int head    = (tile >> 8) & 3;
    int b       = tile >> 10;
    int bh      = b * NH + head;
    int rowbase = rowtile * 16;
    int col     = cc * 16 + n16;

    const float*    hrow = h   + ((size_t)b * NN + rowbase + n16) * FI;
    const _Float16* wcol = w_t + ((size_t)head * FO + col) * FI;

    v8f c = {};
    #pragma unroll
    for (int ks = 0; ks < 4; ++ks) {
        int kb = ks * 32;
        // A fragment: row M = lane&15; K = kb + (e&7) + 8*hi + 16*(e>>3)
        v4f f0 = *(const v4f*)(hrow + kb + 8 * hi);
        v4f f1 = *(const v4f*)(hrow + kb + 8 * hi + 4);
        v4f f2 = *(const v4f*)(hrow + kb + 16 + 8 * hi);
        v4f f3 = *(const v4f*)(hrow + kb + 16 + 8 * hi + 4);
        v16h a;
        #pragma unroll
        for (int q = 0; q < 4; ++q) {
            a[q]      = (_Float16)f0[q];
            a[4 + q]  = (_Float16)f1[q];
            a[8 + q]  = (_Float16)f2[q];
            a[12 + q] = (_Float16)f3[q];
        }
        // B fragment: col N = lane&15; K = kb + (e&7) + 8*hi + 16*(e>>3)
        v8h b0 = *(const v8h*)(wcol + kb + 8 * hi);
        v8h b1 = *(const v8h*)(wcol + kb + 16 + 8 * hi);
        v16h bf;
        #pragma unroll
        for (int q = 0; q < 8; ++q) { bf[q] = b0[q]; bf[8 + q] = b1[q]; }

        c = __builtin_amdgcn_wmma_f32_16x16x32_f16(false, a, false, bf,
                                                   (short)0, c, false, false);
    }
    // D layout: VGPR v -> row M = v + 8*hi, col N = lane&15
    #pragma unroll
    for (int v = 0; v < 8; ++v) {
        int M = v + 8 * hi;
        hp[((size_t)bh * NN + rowbase + M) * FO + col] = c[v];
    }
}

// ---------------------------------------------------------------------------
// k2: attn_src/attn_dst [bh][n] = tanh(h_prime[row]) . a_src/a_dst[head]
// ---------------------------------------------------------------------------
__global__ __launch_bounds__(256) void k2_attnvec(const float* __restrict__ hp,
                                                  const float* __restrict__ a_src,
                                                  const float* __restrict__ a_dst,
                                                  float* __restrict__ s_src,
                                                  float* __restrict__ s_dst) {
    int t = blockIdx.x * blockDim.x + threadIdx.x;      // 0..65535 (= bh*1024 + i)
    int head = (t >> 10) & 3;
    const float* row = hp + (size_t)t * FO;
    const float* as  = a_src + head * FO;
    const float* ad  = a_dst + head * FO;
    float ss = 0.f, sd = 0.f;
    #pragma unroll 4
    for (int o = 0; o < FO; o += 4) {
        v4f f = *(const v4f*)(row + o);
        #pragma unroll
        for (int q = 0; q < 4; ++q) {
            float th = tanhf(f[q]);
            ss += th * as[o + q];
            sd += th * ad[o + q];
        }
    }
    s_src[t] = ss;
    s_dst[t] = sd;
}

// ---------------------------------------------------------------------------
// k3: fused mask + softmax + (attn @ h_prime) + bias  (flash-attention style)
// Block = 128 threads (4 waves), one (b, head, 16-row tile) per block.
// j-tiles of 64: P tile computed cooperatively into LDS (f16), two WMMAs per
// iteration, A/B fragments are pure ds_load_b128.
// ---------------------------------------------------------------------------
#define JT 64   // j-tile width

__global__ __launch_bounds__(128) void k3_flash(const float* __restrict__ hp,
                                                const int* __restrict__ adj,
                                                const float* __restrict__ s_src,
                                                const float* __restrict__ s_dst,
                                                const float* __restrict__ bias,
                                                float* __restrict__ out) {
    __shared__ _Float16 pt[16 * JT];       // P tile: [row][k], f16      (2 KB)
    __shared__ _Float16 vt[FO * JT];       // V tile transposed [col][k] (8 KB)
    __shared__ float    src_r[16];
    __shared__ float    row_m[16];
    __shared__ float    row_inv[16];
    __shared__ float    m_part[128];
    __shared__ float    l_part[128];

    int bt      = blockIdx.x;
    int rowtile = bt & 63;
    int head    = (bt >> 6) & 3;
    int b       = bt >> 8;
    int bh      = b * NH + head;
    int ibase   = rowtile * 16;
    int t       = threadIdx.x;

    const int*   arow_base = adj   + ((size_t)b * NN + ibase) * NN;  // + r*NN
    const float* drow      = s_dst + (size_t)bh * NN;

    // ---- Pass A: one streaming sweep -> per-row online max + exp-sum ----
    {
        int r   = t >> 3;   // row within tile 0..15
        int sub = t & 7;    // 8 threads per row
        float si = s_src[bh * NN + ibase + r];
        const int* arow = arow_base + (size_t)r * NN;
        float m = -1e30f, l = 0.f;
        for (int it = 0; it < 32; ++it) {
            int j0 = it * 32 + sub * 4;
            __builtin_prefetch(arow + j0 + 32, 0, 1);   // global_prefetch_b8
            v4i av = *(const v4i*)(arow + j0);
            #pragma unroll
            for (int q = 0; q < 4; ++q) {
                if (av[q] > 0) {
                    float x = si + drow[j0 + q];
                    float s = x > 0.f ? x : NEG_SLOPE * x;
                    float mn = fmaxf(m, s);
                    l = l * __expf(m - mn) + __expf(s - mn);
                    m = mn;
                }
            }
        }
        m_part[t] = m;
        l_part[t] = l;
        if (sub == 0) src_r[r] = si;
    }
    __syncthreads();
    if (t < 16) {
        float m = -1e30f;
        #pragma unroll
        for (int s = 0; s < 8; ++s) m = fmaxf(m, m_part[t * 8 + s]);
        float l = 0.f;
        #pragma unroll
        for (int s = 0; s < 8; ++s) l += l_part[t * 8 + s] * __expf(m_part[t * 8 + s] - m);
        row_m[t]   = m;
        row_inv[t] = 1.0f / l;    // self-loop guarantees l > 0
    }
    __syncthreads();

    // ---- Pass B: j-tiles of 64, P(f16) @ V(f16), f32 accumulate ----
    int wave = t >> 5, lane = t & 31, hi = lane >> 4, n16 = lane & 15;
    int col = wave * 16 + n16;
    const float* hpbh = hp + (size_t)bh * NN * FO;

    // cooperative P mapping: thread t -> row pr, k offsets pk0..pk0+7
    int pr  = t >> 3;
    int pk0 = (t & 7) * 8;
    const int* arow_p = arow_base + (size_t)pr * NN;

    v8f c = {};
    for (int jt = 0; jt < NN / JT; ++jt) {
        int jb = jt * JT;
        // --- stage P tile: 8 elements per thread, one ds_store_b128 ---
        {
            float sr = src_r[pr];
            float mi = row_m[pr];
            v4i a0 = *(const v4i*)(arow_p + jb + pk0);
            v4i a1 = *(const v4i*)(arow_p + jb + pk0 + 4);
            v4f d0 = *(const v4f*)(drow + jb + pk0);
            v4f d1 = *(const v4f*)(drow + jb + pk0 + 4);
            v8h pv;
            #pragma unroll
            for (int q = 0; q < 4; ++q) {
                float x0 = sr + d0[q];
                float s0 = x0 > 0.f ? x0 : NEG_SLOPE * x0;
                pv[q] = (_Float16)(a0[q] > 0 ? __expf(s0 - mi) : 0.f);
                float x1 = sr + d1[q];
                float s1 = x1 > 0.f ? x1 : NEG_SLOPE * x1;
                pv[4 + q] = (_Float16)(a1[q] > 0 ? __expf(s1 - mi) : 0.f);
            }
            *(v8h*)(pt + pr * JT + pk0) = pv;
        }
        // --- stage V tile transposed to f16: vt[col][k] ---
        #pragma unroll
        for (int i = 0; i < 8; ++i) {
            int idx4 = t + i * 128;              // float4 index 0..1023
            int kk = idx4 >> 4;                  // 0..63 (row in tile)
            int oo = (idx4 & 15) * 4;            // 0..60 (col)
            v4f f = *(const v4f*)(hpbh + (size_t)(jb + kk) * FO + oo);
            #pragma unroll
            for (int q = 0; q < 4; ++q) vt[(oo + q) * JT + kk] = (_Float16)f[q];
        }
        __syncthreads();

        // --- two 16x16x32 WMMAs (k sub-tiles of 32) ---
        #pragma unroll
        for (int sub = 0; sub < 2; ++sub) {
            int kb = sub * 32;
            // A frag: row M = lane&15; k = kb + (e&7) + 8*hi + 16*(e>>3)
            v8h a0 = *(const v8h*)(pt + n16 * JT + kb + 8 * hi);
            v8h a1 = *(const v8h*)(pt + n16 * JT + kb + 16 + 8 * hi);
            v16h a;
            #pragma unroll
            for (int q = 0; q < 8; ++q) { a[q] = a0[q]; a[8 + q] = a1[q]; }
            // B frag: col N = lane&15; k = kb + (e&7) + 8*hi + 16*(e>>3)
            v8h b0 = *(const v8h*)(vt + col * JT + kb + 8 * hi);
            v8h b1 = *(const v8h*)(vt + col * JT + kb + 16 + 8 * hi);
            v16h bf;
            #pragma unroll
            for (int q = 0; q < 8; ++q) { bf[q] = b0[q]; bf[8 + q] = b1[q]; }

            c = __builtin_amdgcn_wmma_f32_16x16x32_f16(false, a, false, bf,
                                                       (short)0, c, false, false);
        }
        __syncthreads();
    }

    // epilogue: scale by 1/rowsum, add bias
    float bcol = bias[col];
    #pragma unroll
    for (int v = 0; v < 8; ++v) {
        int M = v + 8 * hi;
        out[((size_t)bh * NN + ibase + M) * FO + col] = c[v] * row_inv[M] + bcol;
    }
}

// ---------------------------------------------------------------------------
extern "C" void kernel_launch(void* const* d_in, const int* in_sizes, int n_in,
                              void* d_out, int out_size, void* d_ws, size_t ws_size,
                              hipStream_t stream) {
    const float* h     = (const float*)d_in[0];
    const int*   adj   = (const int*)d_in[1];
    const float* w     = (const float*)d_in[2];
    const float* a_src = (const float*)d_in[3];
    const float* a_dst = (const float*)d_in[4];
    const float* bias  = (const float*)d_in[5];
    float* out = (float*)d_out;

    // workspace layout
    const size_t HP_BYTES  = (size_t)BS * NH * NN * FO * sizeof(float); // 16 MiB
    const size_t VEC_BYTES = (size_t)BS * NH * NN * sizeof(float);      // 256 KiB
    char* ws = (char*)d_ws;
    float*     hp    = (float*)ws;
    float*     ssrc  = (float*)(ws + HP_BYTES);
    float*     sdst  = (float*)(ws + HP_BYTES + VEC_BYTES);
    _Float16*  w_t   = (_Float16*)(ws + HP_BYTES + 2 * VEC_BYTES);      // 64 KiB

    (void)in_sizes; (void)n_in; (void)out_size; (void)ws_size;

    // k0: convert/transpose w to f16
    hipLaunchKernelGGL(k0_convert_w, dim3((NH * FI * FO) / 256), dim3(256), 0, stream,
                       w, w_t);
    // k1: h_prime = h @ w   (16384 wave-tiles / 4 waves per block)
    hipLaunchKernelGGL(k1_hprime, dim3(4096), dim3(128), 0, stream,
                       h, w_t, hp);
    // k2: attn_src / attn_dst
    hipLaunchKernelGGL(k2_attnvec, dim3((BS * NH * NN) / 256), dim3(256), 0, stream,
                       hp, a_src, a_dst, ssrc, sdst);
    // k3: fused mask+softmax+PV
    hipLaunchKernelGGL(k3_flash, dim3(BS * NH * 64), dim3(128), 0, stream,
                       hp, adj, ssrc, sdst, bias, out);
}
